// Classifier_63995012711024
// MI455X (gfx1250) — matrix operand order repaired and checked by reading, not compiled, not run on hardware
//
#include <hip/hip_runtime.h>
#include <hip/hip_bf16.h>
#include <math.h>

// Problem constants (match reference)
#define BB   64
#define SS   512
#define IN_  1024
#define EE   128
#define HH   8
#define DD   16
#define NLAYER 4
#define ROWS (BB*SS)   // 32768 rows of the (B,S,E) activations

typedef __attribute__((ext_vector_type(16))) __bf16 v16bf;
typedef __attribute__((ext_vector_type(4)))  __bf16 v4bf;
typedef __attribute__((ext_vector_type(8)))  float  v8f;

// ---------------------------------------------------------------------------
// Tiled GEMM: C(MxN) = A(MxK) * B(KxN) + bias, fp32 in/out, bf16 WMMA compute.
// N and K are compile-time so all strided addresses fold into the 24-bit
// immediate offset field of global_load (single base reg, clauseable loads).
//
// Block = 128 threads (4 wave32s). Block tile 64x64; wave w computes the
// 16x64 strip at rows w*16.. using 4 x v_wmma_f32_16x16x32_bf16 per K-step
// (one shared A fragment, four B fragments). K advances in steps of 32.
//
// fp32 tiles are converted to bf16 ONCE during staging and stored in LDS in
// a per-lane WMMA fragment order (wave32), same contiguous K order for A and B:
//   A 16x32 : lane l -> m=l&15, half=l>>4 ; frag elem j -> K = 16*half + j
//   B 32x16 : lane l -> n=l&15, half=l>>4 ; frag elem j -> K = 16*half + j
//   C 16x16 : lane l -> n=l&15, half=l>>4 ; vgpr r -> M = 8*half + r
// so the hot loop is just contiguous 32B LDS fragment loads + WMMA, and each
// staging thread writes whole 32B fragment rows (packed ds_store_b128).
// RELU=1 applies max(x,0) in the epilogue (FFN first layer).
// ---------------------------------------------------------------------------
template<int RELU, int N, int K>
__global__ __launch_bounds__(128)
void gemm_bf16_wmma(const float* __restrict__ A, const float* __restrict__ Bm,
                    const float* __restrict__ bias, float* __restrict__ C)
{
    // Fragment-ordered bf16 tiles: [strip][lane][fragElem]
    __shared__ alignas(32) __bf16 Abuf[4][32][16];   // 64 rows x 32 K
    __shared__ alignas(32) __bf16 Bbuf[4][32][16];   // 32 K x 64 cols

    const int tid   = threadIdx.x;
    const int lane  = tid & 31;
    const int w     = tid >> 5;        // wave id: M-strip 0..3
    const int m16   = lane & 15;
    const int halfv = lane >> 4;
    const int by    = blockIdx.y;
    const int bx    = blockIdx.x;

    // staging coordinates
    const int ar  = tid >> 1;          // A tile row 0..63
    const int acg = (tid & 1) * 16;    // A tile col group (0 / 16)
    const int bn  = tid & 63;          // B tile col 0..63
    const int bkh = tid >> 6;          // B tile k half (0 / 1)

    const float* aBase = A + (size_t)(by * 64 + ar) * K + acg;
    const float* bBase = Bm + (size_t)(bkh * 16) * N + (bx * 64 + bn);

    v8f acc0 = {}, acc1 = {}, acc2 = {}, acc3 = {};

    for (int k0 = 0; k0 < K; k0 += 32) {
        // ---- stage A (64x32): coalesced float4 loads, cvt->bf16, scatter to
        //      fragment order. Each float4 stays inside one 8-wide K group, so
        //      it lands as 4 consecutive frag elems -> packed 8B LDS stores.
        {
            const float* ap = aBase + k0;
            const int strip = ar >> 4;
            const int am16  = ar & 15;
#pragma unroll
            for (int cc = 0; cc < 16; cc += 4) {
                const int c = acg + cc;            // tile col (K offset)
                float4 f = *(const float4*)(ap + cc);
                const int g  = c >> 4;
                const int hs = (c & 15) >> 3;
                const int u  = c & 7;
                v4bf p;
                p[0] = (__bf16)f.x; p[1] = (__bf16)f.y;
                p[2] = (__bf16)f.z; p[3] = (__bf16)f.w;
                *(v4bf*)&Abuf[strip][g * 16 + am16][hs * 8 + u] = p;
            }
        }
        // ---- stage B (32x64): each thread owns one column n and a 16-deep K
        //      run, so its 16 elements form one contiguous 32B fragment row.
        //      With compile-time N, the 16 coalesced b32 loads are immediate-
        //      offset loads off one base -> 8 packed cvt -> 2 x ds_store_b128.
        {
            const float* bp = bBase + (size_t)k0 * N;
            float fv[16];
#pragma unroll
            for (int j = 0; j < 16; ++j) fv[j] = bp[j * N];
            v16bf pb;
#pragma unroll
            for (int j = 0; j < 16; ++j) pb[j] = (__bf16)fv[j];
            *(v16bf*)&Bbuf[bn >> 4][bkh * 16 + (bn & 15)][0] = pb;
        }
        // prefetch next K-tiles into cache while we compute
        if (k0 + 32 < K) {
            __builtin_prefetch(aBase + (k0 + 32), 0, 1);
            __builtin_prefetch(bBase + (size_t)(k0 + 32) * N, 0, 1);
        }
        __syncthreads();

        // ---- hot loop: contiguous 32B fragment loads + 4 WMMAs per wave
        const v16bf av  = *(const v16bf*)&Abuf[w][lane][0];
        const v16bf bv0 = *(const v16bf*)&Bbuf[0][lane][0];
        const v16bf bv1 = *(const v16bf*)&Bbuf[1][lane][0];
        const v16bf bv2 = *(const v16bf*)&Bbuf[2][lane][0];
        const v16bf bv3 = *(const v16bf*)&Bbuf[3][lane][0];
        acc0 = __builtin_amdgcn_wmma_f32_16x16x32_bf16(false, av, false, bv0,
                                                       (short)0, acc0, false, false);
        acc1 = __builtin_amdgcn_wmma_f32_16x16x32_bf16(false, av, false, bv1,
                                                       (short)0, acc1, false, false);
        acc2 = __builtin_amdgcn_wmma_f32_16x16x32_bf16(false, av, false, bv2,
                                                       (short)0, acc2, false, false);
        acc3 = __builtin_amdgcn_wmma_f32_16x16x32_bf16(false, av, false, bv3,
                                                       (short)0, acc3, false, false);
        __syncthreads();
    }

    // ---- epilogue: bias (+ optional relu), write fp32
    const int row0 = by * 64 + w * 16 + halfv * 8;
#pragma unroll
    for (int t = 0; t < 4; ++t) {
        const v8f* accp = (t == 0) ? &acc0 : (t == 1) ? &acc1 : (t == 2) ? &acc2 : &acc3;
        const int n = bx * 64 + t * 16 + m16;
        const float bsv = bias[n];
#pragma unroll
        for (int r = 0; r < 8; ++r) {
            float val = (*accp)[r] + bsv;
            if (RELU) val = fmaxf(val, 0.0f);
            C[(size_t)(row0 + r) * N + n] = val;
        }
    }
}

// ---------------------------------------------------------------------------
// Attention across recordings: per (s,h) a 64x64 softmax(QK^T/4)V with D=16.
// One 64-thread block per (s,h); K/V rows staged in LDS; thread i owns query
// row i. Two-pass (max, then exp/accumulate) softmax, -inf on padded keys.
// ---------------------------------------------------------------------------
__global__ __launch_bounds__(64)
void attn_kernel(const float* __restrict__ q, const float* __restrict__ k,
                 const float* __restrict__ v, const unsigned char* __restrict__ mask,
                 float* __restrict__ o)
{
    const int s  = blockIdx.x >> 3;   // / HH
    const int hh = blockIdx.x & 7;    // % HH
    const int i  = threadIdx.x;       // recording index 0..63

    __shared__ float ks[BB][DD + 1];
    __shared__ float vs[BB][DD + 1];
    __shared__ float msk[BB];

    const size_t base = ((size_t)i * SS + s) * EE + hh * DD;
    {
        const float4* kp = (const float4*)(k + base);
        const float4* vp = (const float4*)(v + base);
#pragma unroll
        for (int t = 0; t < 4; ++t) {
            float4 kv = kp[t], vv4 = vp[t];
            ks[i][t * 4 + 0] = kv.x;  ks[i][t * 4 + 1] = kv.y;
            ks[i][t * 4 + 2] = kv.z;  ks[i][t * 4 + 3] = kv.w;
            vs[i][t * 4 + 0] = vv4.x; vs[i][t * 4 + 1] = vv4.y;
            vs[i][t * 4 + 2] = vv4.z; vs[i][t * 4 + 3] = vv4.w;
        }
        msk[i] = mask[(size_t)i * SS + s] ? 1.0f : 0.0f;
    }
    __syncthreads();

    float qi[DD];
#pragma unroll
    for (int t = 0; t < 4; ++t) {
        float4 qv = *(const float4*)(q + base + t * 4);
        qi[t * 4 + 0] = qv.x; qi[t * 4 + 1] = qv.y;
        qi[t * 4 + 2] = qv.z; qi[t * 4 + 3] = qv.w;
    }

    const float scale = 0.25f;  // 1/sqrt(16)
    float mmax = -3.0e38f;
    for (int j = 0; j < BB; ++j) {
        float sc = 0.0f;
#pragma unroll
        for (int d = 0; d < DD; ++d) sc += qi[d] * ks[j][d];
        sc = (msk[j] > 0.5f) ? -3.0e38f : sc * scale;
        mmax = fmaxf(mmax, sc);
    }

    float ssum = 0.0f;
    float accv[DD] = {};
    for (int j = 0; j < BB; ++j) {
        float sc = 0.0f;
#pragma unroll
        for (int d = 0; d < DD; ++d) sc += qi[d] * ks[j][d];
        sc = (msk[j] > 0.5f) ? -3.0e38f : sc * scale;
        float p = __expf(sc - mmax);
        ssum += p;
#pragma unroll
        for (int d = 0; d < DD; ++d) accv[d] += p * vs[j][d];
    }
    const float inv = 1.0f / ssum;
#pragma unroll
    for (int d = 0; d < DD; ++d) o[base + d] = accv[d] * inv;
}

// ---------------------------------------------------------------------------
// LayerNorm(a + res) * g + b, E=128. One wave32 per row, 4 elems per lane,
// wave-level shuffle reductions. Safe when out aliases res (each address is
// read and written by exactly one thread, read-before-write).
// ---------------------------------------------------------------------------
__global__ __launch_bounds__(256)
void ln_kernel(const float* __restrict__ a, const float* __restrict__ res,
               const float* __restrict__ g, const float* __restrict__ b,
               float* __restrict__ out)
{
    const int row  = blockIdx.x * 8 + (threadIdx.x >> 5);
    const int lane = threadIdx.x & 31;
    const size_t base = (size_t)row * EE;

    float x[4];
    float s1 = 0.0f, s2 = 0.0f;
#pragma unroll
    for (int t = 0; t < 4; ++t) {
        int e = lane + t * 32;
        x[t] = a[base + e] + res[base + e];
        s1 += x[t];
        s2 += x[t] * x[t];
    }
#pragma unroll
    for (int off = 16; off > 0; off >>= 1) {
        s1 += __shfl_xor(s1, off);
        s2 += __shfl_xor(s2, off);
    }
    const float mean = s1 * (1.0f / EE);
    const float var  = s2 * (1.0f / EE) - mean * mean;
    const float inv  = rsqrtf(var + 1e-5f);
#pragma unroll
    for (int t = 0; t < 4; ++t) {
        int e = lane + t * 32;
        out[base + e] = (x[t] - mean) * inv * g[e] + b[e];
    }
}

// Sinusoidal positional encoding added in place (re-added every block, as in
// the reference source).
__global__ void addpe_kernel(float* __restrict__ h)
{
    const int idx = blockIdx.x * blockDim.x + threadIdx.x;
    const int e = idx & (EE - 1);
    const int s = (idx >> 7) & (SS - 1);   // row % S
    const float expo  = (float)(2 * (e >> 1)) * (1.0f / EE);
    const float denom = __powf(10000.0f, expo);
    const float ang   = (float)s / denom;
    h[idx] += (e & 1) ? __cosf(ang) : __sinf(ang);
}

// Masked mean pool over epochs: one block per recording, one thread per chan.
__global__ __launch_bounds__(128)
void pool_kernel(const float* __restrict__ h, const unsigned char* __restrict__ mask,
                 float* __restrict__ pooled)
{
    const int b = blockIdx.x, e = threadIdx.x;
    float sum = 0.0f, cnt = 0.0f;
    for (int s = 0; s < SS; ++s) {
        float valid = mask[(size_t)b * SS + s] ? 0.0f : 1.0f;
        sum += valid * h[((size_t)b * SS + s) * EE + e];
        cnt += valid;
    }
    pooled[b * EE + e] = sum / cnt;
}

// fc1(128->32)+relu, fc2(32->1), sigmoid. One thread per recording.
__global__ __launch_bounds__(64)
void head_kernel(const float* __restrict__ pooled,
                 const float* __restrict__ fc1w, const float* __restrict__ fc1b,
                 const float* __restrict__ fc2w, const float* __restrict__ fc2b,
                 float* __restrict__ out)
{
    const int b = threadIdx.x;
    float acc2 = fc2b[0];
    for (int oo = 0; oo < 32; ++oo) {
        float t = fc1b[oo];
        for (int i2 = 0; i2 < EE; ++i2) t += pooled[b * EE + i2] * fc1w[i2 * 32 + oo];
        t = fmaxf(t, 0.0f);
        acc2 += t * fc2w[oo];
    }
    out[b] = 1.0f / (1.0f + __expf(-acc2));
}

// ---------------------------------------------------------------------------
extern "C" void kernel_launch(void* const* d_in, const int* in_sizes, int n_in,
                              void* d_out, int out_size, void* d_ws, size_t ws_size,
                              hipStream_t stream)
{
    (void)in_sizes; (void)n_in; (void)out_size; (void)ws_size;

    const float*         x       = (const float*)d_in[0];
    const unsigned char* mask    = (const unsigned char*)d_in[1];  // jnp bool_, 1 byte/elem
    const float*         embed_w = (const float*)d_in[2];
    const float*         embed_b = (const float*)d_in[3];
    const float*         qkv_w   = (const float*)d_in[4];
    const float*         qkv_b   = (const float*)d_in[5];
    const float*         out_w   = (const float*)d_in[6];
    const float*         out_b   = (const float*)d_in[7];
    const float*         ln_g    = (const float*)d_in[8];
    const float*         ln_b    = (const float*)d_in[9];
    const float*         ff1_w   = (const float*)d_in[10];
    const float*         ff1_b   = (const float*)d_in[11];
    const float*         ff2_w   = (const float*)d_in[12];
    const float*         ff2_b   = (const float*)d_in[13];
    const float*         fc1_w   = (const float*)d_in[14];
    const float*         fc1_b   = (const float*)d_in[15];
    const float*         fc2_w   = (const float*)d_in[16];
    const float*         fc2_b   = (const float*)d_in[17];

    // Workspace layout (peak ~96 MB, fp32 everywhere):
    char*  ws  = (char*)d_ws;
    float* h   = (float*)(ws);                               // 16 MB activations
    float* qb_ = (float*)(ws + (size_t)16 * 1024 * 1024);    // 16 MB Q
    float* kb_ = (float*)(ws + (size_t)32 * 1024 * 1024);    // 16 MB K
    float* vb_ = (float*)(ws + (size_t)48 * 1024 * 1024);    // 16 MB V
    float* t0  = (float*)(ws + (size_t)64 * 1024 * 1024);    // 16 MB attn proj out
    float* f1  = (float*)(ws + (size_t)16 * 1024 * 1024);    // 64 MB ff1 out (reuses q/k/v/t0)
    float* o80 = (float*)(ws + (size_t)80 * 1024 * 1024);    // 16 MB attn raw out / ff2 out
    float* pooled = (float*)(ws + (size_t)96 * 1024 * 1024); // 32 KB

    const dim3 blk(128);
    const dim3 gE(EE / 64, ROWS / 64);        // N=128 GEMMs: 2 x 512 blocks
    const dim3 gF((4 * EE) / 64, ROWS / 64);  // N=512 GEMM (ff1): 8 x 512 blocks

    // h = x @ embed_w + embed_b
    gemm_bf16_wmma<0, EE, IN_><<<gE, blk, 0, stream>>>(x, embed_w, embed_b, h);

    for (int l = 0; l < NLAYER; ++l) {
        addpe_kernel<<<(ROWS * EE) / 256, 256, 0, stream>>>(h);

        const float* wq = qkv_w + ((size_t)l * 3 + 0) * EE * EE;
        const float* wk = qkv_w + ((size_t)l * 3 + 1) * EE * EE;
        const float* wv = qkv_w + ((size_t)l * 3 + 2) * EE * EE;
        const float* bq = qkv_b + ((size_t)l * 3 + 0) * EE;
        const float* bk = qkv_b + ((size_t)l * 3 + 1) * EE;
        const float* bv = qkv_b + ((size_t)l * 3 + 2) * EE;

        gemm_bf16_wmma<0, EE, EE><<<gE, blk, 0, stream>>>(h, wq, bq, qb_);
        gemm_bf16_wmma<0, EE, EE><<<gE, blk, 0, stream>>>(h, wk, bk, kb_);
        gemm_bf16_wmma<0, EE, EE><<<gE, blk, 0, stream>>>(h, wv, bv, vb_);

        attn_kernel<<<SS * HH, 64, 0, stream>>>(qb_, kb_, vb_, mask, o80);

        gemm_bf16_wmma<0, EE, EE><<<gE, blk, 0, stream>>>(o80, out_w + (size_t)l * EE * EE,
                                                          out_b + (size_t)l * EE, t0);
        ln_kernel<<<ROWS / 8, 256, 0, stream>>>(t0, h, ln_g + (size_t)l * EE,
                                                ln_b + (size_t)l * EE, h);

        gemm_bf16_wmma<1, 4 * EE, EE><<<gF, blk, 0, stream>>>(h, ff1_w + (size_t)l * EE * 4 * EE,
                                                              ff1_b + (size_t)l * 4 * EE, f1);
        gemm_bf16_wmma<0, EE, 4 * EE><<<gE, blk, 0, stream>>>(f1, ff2_w + (size_t)l * 4 * EE * EE,
                                                              ff2_b + (size_t)l * EE, o80);
        ln_kernel<<<ROWS / 8, 256, 0, stream>>>(o80, h, ln_g + (size_t)l * EE,
                                                ln_b + (size_t)l * EE, h);
    }

    pool_kernel<<<BB, EE, 0, stream>>>(h, mask, pooled);
    head_kernel<<<1, BB, 0, stream>>>(pooled, fc1_w, fc1_b, fc2_w, fc2_b, (float*)d_out);
}
